// QuantumNeuralNetwork_55980603736431
// MI455X (gfx1250) — matrix operand order repaired and checked by reading, not compiled, not run on hardware
//
#include <hip/hip_runtime.h>
#include <math.h>

typedef __attribute__((ext_vector_type(16))) _Float16 v16h;
typedef __attribute__((ext_vector_type(8)))  float    v8f;

#define IN_DIM 512
#define HID    1024
#define NQ     8
#define NL     4

struct c2 { float x, y; };
__device__ inline c2 cmul(c2 a, c2 b) { return { a.x*b.x - a.y*b.y, a.x*b.y + a.y*b.x }; }
__device__ inline c2 cmac(c2 g, c2 a, c2 acc) {   // acc + g*a
  acc.x = fmaf(g.x, a.x, fmaf(-g.y, a.y, acc.x));
  acc.y = fmaf(g.x, a.y, fmaf( g.y, a.x, acc.y));
  return acc;
}
__device__ inline c2 shfl2(c2 v, int mask) {
  c2 r; r.x = __shfl_xor(v.x, mask, 32); r.y = __shfl_xor(v.y, mask, 32); return r;
}

__global__ __launch_bounds__(256)
void qnn_fused_kernel(const float* __restrict__ x,  const float* __restrict__ W1,
                      const float* __restrict__ b1, const float* __restrict__ W2,
                      const float* __restrict__ b2, const float* __restrict__ qw,
                      const float* __restrict__ W3, const float* __restrict__ b3,
                      float* __restrict__ out)
{
  __shared__ _Float16 hs[16 * HID];               // 32 KB h tile (16 samples x 1024, f16)
  __shared__ float    gm[NL * NQ * 8];            // 1 KB: 32 Rot gates, 4 complex each

  const int tid  = threadIdx.x;
  const int lane = tid & 31;
  const int wave = tid >> 5;
  const int g    = lane >> 4;                     // half-wave select
  const int mn   = lane & 15;                     // A row / B,C column within tile
  const int tileM = blockIdx.x;                   // 16 samples per block
  const int rowA  = tileM * 16 + mn;

  // ---- precompute shared Rot gate matrices (one gate per thread, 32 threads) ----
  if (tid < NL * NQ) {
    const float phi = qw[tid * 4 + 0];
    const float th  = qw[tid * 4 + 1];
    const float om  = qw[tid * 4 + 2];
    const float ct = __cosf(th * 0.5f),         st = __sinf(th * 0.5f);
    const float cp = __cosf((phi + om) * 0.5f), sp = __sinf((phi + om) * 0.5f);
    const float cm = __cosf((phi - om) * 0.5f), sm = __sinf((phi - om) * 0.5f);
    float* gp = gm + tid * 8;
    gp[0] =  cp * ct; gp[1] = -sp * ct;           // g00 =  e^{-i(phi+om)/2} cos
    gp[2] = -cm * st; gp[3] = -sm * st;           // g01 = -e^{+i(phi-om)/2} sin
    gp[4] =  cm * st; gp[5] = -sm * st;           // g10 =  e^{-i(phi-om)/2} sin
    gp[6] =  cp * ct; gp[7] =  sp * ct;           // g11 =  e^{+i(phi+om)/2} cos
  }

  // ================= Phase 1: h = relu(x @ W1^T + b1), WMMA f16 =================
  v8f acc[8];
  #pragma unroll
  for (int t = 0; t < 8; ++t) acc[t] = (v8f){};

  const float* xrow = x + (size_t)rowA * IN_DIM;
  for (int ks = 0; ks < IN_DIM / 32; ++ks) {
    const int K0 = ks * 32;
    // A fragment (16x32 f16): lane(l<16): K in {0..15}; lane(l>=16): +16
    v16h afrag;
    #pragma unroll
    for (int v = 0; v < 8; ++v) {
      const int kk = v * 2 + g * 16;
      const float2 t = *reinterpret_cast<const float2*>(xrow + K0 + kk);
      afrag[2*v]   = (_Float16)t.x;
      afrag[2*v+1] = (_Float16)t.y;
    }
    #pragma unroll
    for (int sub = 0; sub < 8; ++sub) {
      const int col = wave * 128 + sub * 16 + mn;           // column of W1 (= N)
      const float* wrow = W1 + (size_t)col * IN_DIM + K0 + g * 16;
      v16h bfrag;                                           // B: half-wave K = g*16 + j
      #pragma unroll
      for (int v = 0; v < 8; ++v) {
        const float2 t = *reinterpret_cast<const float2*>(wrow + 2 * v);
        bfrag[2*v]   = (_Float16)t.x;
        bfrag[2*v+1] = (_Float16)t.y;
      }
      acc[sub] = __builtin_amdgcn_wmma_f32_16x16x32_f16(
          false, afrag, false, bfrag, (short)0, acc[sub], false, false);
    }
  }
  #pragma unroll
  for (int sub = 0; sub < 8; ++sub) {
    const int colBase = wave * 128 + sub * 16;
    const float bias = b1[colBase + mn];
    #pragma unroll
    for (int r = 0; r < 8; ++r)                             // D: M = r + 8*g, N = mn
      hs[(r + 8 * g) * HID + colBase + mn] = (_Float16)fmaxf(acc[sub][r] + bias, 0.0f);
  }
  __syncthreads();

  // ============ Phase 2: per-sample GEMM2 + 8-qubit circuit + GEMM3 =============
  for (int rep = 0; rep < 2; ++rep) {
    const int sLocal = wave * 2 + rep;
    const int sample = tileM * 16 + sLocal;
    const _Float16* hrow = hs + sLocal * HID;

    // a = relu(h @ W2^T + b2): 8 dots of length 1024, lane-strided
    float ang[NQ];
    #pragma unroll
    for (int q = 0; q < NQ; ++q) ang[q] = 0.0f;
    for (int i = lane; i < HID; i += 32) {
      const float hv = (float)hrow[i];
      #pragma unroll
      for (int q = 0; q < NQ; ++q) ang[q] = fmaf(hv, W2[q * HID + i], ang[q]);
    }
    #pragma unroll
    for (int q = 0; q < NQ; ++q) {
      float v = ang[q];
      #pragma unroll
      for (int off = 16; off > 0; off >>= 1) v += __shfl_xor(v, off, 32);
      ang[q] = fmaxf(v + b2[q], 0.0f);                      // RY angle
    }

    // ---- initial state: RY(ang) layer applied to |0..0> is a real product state ----
    // amplitude i = k*32 + lane; wires 0..2 -> k bits 2..0, wires 3..7 -> lane bits 4..0
    float cs[NQ], sn[NQ];
    #pragma unroll
    for (int w = 0; w < NQ; ++w) {
      cs[w] = __cosf(ang[w] * 0.5f);
      sn[w] = __sinf(ang[w] * 0.5f);
    }
    const float flane = ((lane & 16) ? sn[3] : cs[3]) *
                        ((lane &  8) ? sn[4] : cs[4]) *
                        ((lane &  4) ? sn[5] : cs[5]) *
                        ((lane &  2) ? sn[6] : cs[6]) *
                        ((lane &  1) ? sn[7] : cs[7]);
    c2 amp[8];
    #pragma unroll
    for (int k = 0; k < 8; ++k) {
      amp[k].x = flane * ((k & 4) ? sn[0] : cs[0])
                       * ((k & 2) ? sn[1] : cs[1])
                       * ((k & 1) ? sn[2] : cs[2]);
      amp[k].y = 0.0f;
    }

    for (int l = 0; l < NL; ++l) {
      // ---- CNOT ring (0,1)(1,2)(2,3)(3,4)(4,5)(5,6)(6,7)(7,0) ----
      { c2 t = amp[4]; amp[4] = amp[6]; amp[6] = t;         // (0,1): k&4 -> swap k^2
        t = amp[5]; amp[5] = amp[7]; amp[7] = t; }
      { c2 t = amp[2]; amp[2] = amp[3]; amp[3] = t;         // (1,2): k&2 -> swap k^1
        t = amp[6]; amp[6] = amp[7]; amp[7] = t; }
      #pragma unroll
      for (int k = 1; k < 8; k += 2)                        // (2,3): k&1 -> lane swap 16
        amp[k] = shfl2(amp[k], 16);
      #pragma unroll
      for (int k = 0; k < 8; ++k) {                         // (3,4): lane&16 -> lane swap 8
        const c2 p = shfl2(amp[k], 8);
        if (lane & 16) amp[k] = p;
      }
      #pragma unroll
      for (int k = 0; k < 8; ++k) {                         // (4,5)
        const c2 p = shfl2(amp[k], 4);
        if (lane & 8) amp[k] = p;
      }
      #pragma unroll
      for (int k = 0; k < 8; ++k) {                         // (5,6)
        const c2 p = shfl2(amp[k], 2);
        if (lane & 4) amp[k] = p;
      }
      #pragma unroll
      for (int k = 0; k < 8; ++k) {                         // (6,7)
        const c2 p = shfl2(amp[k], 1);
        if (lane & 2) amp[k] = p;
      }
      {                                                     // (7,0): lane&1 -> swap k^4
        const bool c70 = (lane & 1) != 0;
        #pragma unroll
        for (int k = 0; k < 4; ++k) {
          const c2 lo = amp[k], hi = amp[k + 4];
          amp[k]     = c70 ? hi : lo;
          amp[k + 4] = c70 ? lo : hi;
        }
      }
      // ---- shared Rot gates (precomputed in LDS) ----
      #pragma unroll
      for (int w = 0; w < 8; ++w) {
        const float* gp = gm + (l * NQ + w) * 8;
        const c2 g00 = { gp[0], gp[1] };
        const c2 g01 = { gp[2], gp[3] };
        const c2 g10 = { gp[4], gp[5] };
        const c2 g11 = { gp[6], gp[7] };
        if (w < 3) {
          const int km = 4 >> w;
          #pragma unroll
          for (int k = 0; k < 8; ++k) if (!(k & km)) {
            const c2 a0 = amp[k], a1 = amp[k | km];
            amp[k]      = cmac(g01, a1, cmul(g00, a0));
            amp[k | km] = cmac(g11, a1, cmul(g10, a0));
          }
        } else {
          const int lm = 16 >> (w - 3);
          const bool hi = (lane & lm) != 0;
          const c2 ca = hi ? g11 : g00;                     // coeff of my amplitude
          const c2 cb = hi ? g10 : g01;                     // coeff of partner
          #pragma unroll
          for (int k = 0; k < 8; ++k) {
            const c2 p = shfl2(amp[k], lm);
            amp[k] = cmac(cb, p, cmul(ca, amp[k]));
          }
        }
      }
    }

    // ---- <Z_w> = sum_i sign_w(i) |amp_i|^2 ----
    float pk[8], tot = 0.0f;
    #pragma unroll
    for (int k = 0; k < 8; ++k) {
      pk[k] = fmaf(amp[k].x, amp[k].x, amp[k].y * amp[k].y);
      tot += pk[k];
    }
    float z[8];
    z[0] = z[1] = z[2] = 0.0f;
    #pragma unroll
    for (int k = 0; k < 8; ++k) {
      z[0] += (k & 4) ? -pk[k] : pk[k];
      z[1] += (k & 2) ? -pk[k] : pk[k];
      z[2] += (k & 1) ? -pk[k] : pk[k];
    }
    z[3] = (lane & 16) ? -tot : tot;
    z[4] = (lane & 8)  ? -tot : tot;
    z[5] = (lane & 4)  ? -tot : tot;
    z[6] = (lane & 2)  ? -tot : tot;
    z[7] = (lane & 1)  ? -tot : tot;
    #pragma unroll
    for (int w = 0; w < 8; ++w) {
      #pragma unroll
      for (int off = 16; off > 0; off >>= 1) z[w] += __shfl_xor(z[w], off, 32);
    }

    if (lane == 0) {                                        // out = z @ W3^T + b3
      float o0 = b3[0], o1 = b3[1];
      #pragma unroll
      for (int w = 0; w < 8; ++w) {
        o0 = fmaf(z[w], W3[w], o0);
        o1 = fmaf(z[w], W3[8 + w], o1);
      }
      out[sample * 2 + 0] = o0;
      out[sample * 2 + 1] = o1;
    }
  }
}

extern "C" void kernel_launch(void* const* d_in, const int* in_sizes, int n_in,
                              void* d_out, int out_size, void* d_ws, size_t ws_size,
                              hipStream_t stream) {
  (void)in_sizes; (void)n_in; (void)d_ws; (void)ws_size; (void)out_size;
  const float* x  = (const float*)d_in[0];
  const float* W1 = (const float*)d_in[1];
  const float* b1 = (const float*)d_in[2];
  const float* W2 = (const float*)d_in[3];
  const float* b2 = (const float*)d_in[4];
  const float* qw = (const float*)d_in[5];
  const float* W3 = (const float*)d_in[6];
  const float* b3 = (const float*)d_in[7];
  float* out = (float*)d_out;
  // 16384 samples / 16 per block = 1024 blocks; 256 threads = 8 wave32
  qnn_fused_kernel<<<dim3(1024), dim3(256), 0, stream>>>(x, W1, b1, W2, b2, qw, W3, b3, out);
}
